// ConvAttnState_59966333386902
// MI455X (gfx1250) — matrix operand order repaired and checked by reading, not compiled, not run on hardware
//
#include <hip/hip_runtime.h>

// ---------------------------------------------------------------------------
// Problem dims
// ---------------------------------------------------------------------------
enum { DB = 8, DH = 12, DL = 2048, DD = 64, DE = 768, DLQ = 1024 };
// rows
enum { NROW = DB * DL, NQROW = DB * DLQ };  // 16384, 8192

typedef __attribute__((ext_vector_type(16))) __bf16 v16bf;
typedef __attribute__((ext_vector_type(8)))  float  v8f;

union FragBF {
    v16bf v;
    uint4 q[2];
    unsigned short h[16];
};

__device__ __forceinline__ unsigned short f32_bf16_rne(float f) {
    unsigned int u = __float_as_uint(f);
    u += 0x7FFFu + ((u >> 16) & 1u);
    return (unsigned short)(u >> 16);
}

__device__ __forceinline__ v8f wmma_bf16(v16bf a, v16bf b, v8f c) {
    return __builtin_amdgcn_wmma_f32_16x16x32_bf16(false, a, false, b,
                                                   (short)0, c, false, false);
}

// ---------------------------------------------------------------------------
// Elementwise prep kernels
// ---------------------------------------------------------------------------

// state [B,H,L,D] f32 -> x_bf [B*L, E] bf16   (e = h*64 + d)
__global__ __launch_bounds__(256) void k_permute_bf16(const float* __restrict__ state,
                                                      unsigned short* __restrict__ xbf) {
    int idx = blockIdx.x * 256 + threadIdx.x;
    if (idx >= NROW * DE) return;
    int e = idx % DE;
    int l = (idx / DE) % DL;
    int b = idx / (DE * DL);
    int h = e >> 6, d = e & 63;
    float v = state[(((size_t)(b * DH + h)) * DL + l) * DD + d];
    xbf[idx] = f32_bf16_rne(v);
}

__global__ __launch_bounds__(256) void k_cvt_bf16(const float* __restrict__ s,
                                                  unsigned short* __restrict__ d, int n) {
    int idx = blockIdx.x * 256 + threadIdx.x;
    if (idx < n) d[idx] = f32_bf16_rne(s[idx]);
}

// Wq [t][i][o] f32 -> Wq_bf [t][o][i] bf16
__global__ __launch_bounds__(256) void k_transpose_wq(const float* __restrict__ s,
                                                      unsigned short* __restrict__ d) {
    int idx = blockIdx.x * 256 + threadIdx.x;
    if (idx >= 3 * DE * DE) return;
    int t = idx / (DE * DE);
    int r = idx % (DE * DE);
    int i = r / DE, o = r % DE;
    d[(size_t)t * DE * DE + (size_t)o * DE + i] = f32_bf16_rne(s[idx]);
}

// ---------------------------------------------------------------------------
// Generic GEMM:  C[M,768] = act(A[M,768] @ W^T + bias)
// A bf16 row-major, W bf16 row-major [n][k] (== B column-major-over-K).
// Block: 128 threads (4 waves); each wave computes a 64x64 tile (4x4 WMMA frags).
// grid = (3, M/64)
// ---------------------------------------------------------------------------
__global__ __launch_bounds__(128) void k_gemm_rm(const unsigned short* __restrict__ A,
                                                 const unsigned short* __restrict__ W,
                                                 const float* __restrict__ bias,
                                                 unsigned short* __restrict__ C,
                                                 int relu) {
    int wave = threadIdx.x >> 5, lane = threadIdx.x & 31;
    int mlane = lane & 15, hl = (lane >> 4) & 1;
    int m0 = blockIdx.y * 64;
    int n0 = blockIdx.x * 256 + wave * 64;

    v8f acc[4][4];
#pragma unroll
    for (int i = 0; i < 4; ++i)
#pragma unroll
        for (int j = 0; j < 4; ++j)
#pragma unroll
            for (int r = 0; r < 8; ++r) acc[i][j][r] = 0.0f;

    for (int kb = 0; kb < DE; kb += 32) {
        FragBF fa[4], fb[4];
#pragma unroll
        for (int i = 0; i < 4; ++i) {
            const unsigned short* p = A + ((size_t)(m0 + i * 16 + mlane)) * DE + kb + hl * 8;
            fa[i].q[0] = *reinterpret_cast<const uint4*>(p);
            fa[i].q[1] = *reinterpret_cast<const uint4*>(p + 16);
        }
#pragma unroll
        for (int j = 0; j < 4; ++j) {
            const unsigned short* p = W + ((size_t)(n0 + j * 16 + mlane)) * DE + kb + hl * 8;
            fb[j].q[0] = *reinterpret_cast<const uint4*>(p);
            fb[j].q[1] = *reinterpret_cast<const uint4*>(p + 16);
        }
#pragma unroll
        for (int i = 0; i < 4; ++i)
#pragma unroll
            for (int j = 0; j < 4; ++j)
                acc[i][j] = wmma_bf16(fa[i].v, fb[j].v, acc[i][j]);
    }

#pragma unroll
    for (int i = 0; i < 4; ++i)
#pragma unroll
        for (int j = 0; j < 4; ++j) {
            int n = n0 + j * 16 + mlane;
            float bs = bias[n];
#pragma unroll
            for (int r = 0; r < 8; ++r) {
                int m = m0 + i * 16 + hl * 8 + r;
                float v = acc[i][j][r] + bs;
                if (relu) v = fmaxf(v, 0.0f);
                C[(size_t)m * DE + n] = f32_bf16_rne(v);
            }
        }
}

// ---------------------------------------------------------------------------
// Strided conv query projection as 3 accumulated GEMMs.
// Q[b,lq,o] = sum_t sum_i X[b, 2*lq+t-1, i] * Wq[t][o][i]  (+bq)
// M = 8192 (B*Lq). grid = (3, 128).
// ---------------------------------------------------------------------------
__global__ __launch_bounds__(128) void k_gemm_conv(const unsigned short* __restrict__ X,
                                                   const unsigned short* __restrict__ Wq3,
                                                   const float* __restrict__ bias,
                                                   unsigned short* __restrict__ Q) {
    int wave = threadIdx.x >> 5, lane = threadIdx.x & 31;
    int mlane = lane & 15, hl = (lane >> 4) & 1;
    int m0 = blockIdx.y * 64;
    int n0 = blockIdx.x * 256 + wave * 64;

    v8f acc[4][4];
#pragma unroll
    for (int i = 0; i < 4; ++i)
#pragma unroll
        for (int j = 0; j < 4; ++j)
#pragma unroll
            for (int r = 0; r < 8; ++r) acc[i][j][r] = 0.0f;

    for (int t = 0; t < 3; ++t) {
        const unsigned short* Wt = Wq3 + (size_t)t * DE * DE;
        for (int kb = 0; kb < DE; kb += 32) {
            FragBF fa[4], fb[4];
#pragma unroll
            for (int i = 0; i < 4; ++i) {
                int mg = m0 + i * 16 + mlane;      // global q-row
                int b = mg >> 10;
                int lq = mg & (DLQ - 1);
                int l = 2 * lq + t - 1;
                if (l >= 0 && l < DL) {
                    const unsigned short* p =
                        X + ((size_t)(b * DL + l)) * DE + kb + hl * 8;
                    fa[i].q[0] = *reinterpret_cast<const uint4*>(p);
                    fa[i].q[1] = *reinterpret_cast<const uint4*>(p + 16);
                } else {
                    fa[i].q[0] = make_uint4(0u, 0u, 0u, 0u);
                    fa[i].q[1] = make_uint4(0u, 0u, 0u, 0u);
                }
            }
#pragma unroll
            for (int j = 0; j < 4; ++j) {
                const unsigned short* p =
                    Wt + ((size_t)(n0 + j * 16 + mlane)) * DE + kb + hl * 8;
                fb[j].q[0] = *reinterpret_cast<const uint4*>(p);
                fb[j].q[1] = *reinterpret_cast<const uint4*>(p + 16);
            }
#pragma unroll
            for (int i = 0; i < 4; ++i)
#pragma unroll
                for (int j = 0; j < 4; ++j)
                    acc[i][j] = wmma_bf16(fa[i].v, fb[j].v, acc[i][j]);
        }
    }

#pragma unroll
    for (int i = 0; i < 4; ++i)
#pragma unroll
        for (int j = 0; j < 4; ++j) {
            int n = n0 + j * 16 + mlane;
            float bs = bias[n];
#pragma unroll
            for (int r = 0; r < 8; ++r) {
                int m = m0 + i * 16 + hl * 8 + r;
                Q[(size_t)m * DE + n] = f32_bf16_rne(acc[i][j][r] + bs);
            }
        }
}

// ---------------------------------------------------------------------------
// V projection with transposed store: vT[b][h][d][l]  (K-contiguous for P·V)
// ---------------------------------------------------------------------------
__global__ __launch_bounds__(128) void k_gemm_vT(const unsigned short* __restrict__ A,
                                                 const unsigned short* __restrict__ W,
                                                 const float* __restrict__ bias,
                                                 unsigned short* __restrict__ VT) {
    int wave = threadIdx.x >> 5, lane = threadIdx.x & 31;
    int mlane = lane & 15, hl = (lane >> 4) & 1;
    int m0 = blockIdx.y * 64;
    int n0 = blockIdx.x * 256 + wave * 64;

    v8f acc[4][4];
#pragma unroll
    for (int i = 0; i < 4; ++i)
#pragma unroll
        for (int j = 0; j < 4; ++j)
#pragma unroll
            for (int r = 0; r < 8; ++r) acc[i][j][r] = 0.0f;

    for (int kb = 0; kb < DE; kb += 32) {
        FragBF fa[4], fb[4];
#pragma unroll
        for (int i = 0; i < 4; ++i) {
            const unsigned short* p = A + ((size_t)(m0 + i * 16 + mlane)) * DE + kb + hl * 8;
            fa[i].q[0] = *reinterpret_cast<const uint4*>(p);
            fa[i].q[1] = *reinterpret_cast<const uint4*>(p + 16);
        }
#pragma unroll
        for (int j = 0; j < 4; ++j) {
            const unsigned short* p = W + ((size_t)(n0 + j * 16 + mlane)) * DE + kb + hl * 8;
            fb[j].q[0] = *reinterpret_cast<const uint4*>(p);
            fb[j].q[1] = *reinterpret_cast<const uint4*>(p + 16);
        }
#pragma unroll
        for (int i = 0; i < 4; ++i)
#pragma unroll
            for (int j = 0; j < 4; ++j)
                acc[i][j] = wmma_bf16(fa[i].v, fb[j].v, acc[i][j]);
    }

#pragma unroll
    for (int i = 0; i < 4; ++i)
#pragma unroll
        for (int j = 0; j < 4; ++j) {
            int n = n0 + j * 16 + mlane;          // feature e
            int h = n >> 6, d = n & 63;
            float bs = bias[n];
#pragma unroll
            for (int r = 0; r < 8; ++r) {
                int mg = m0 + i * 16 + hl * 8 + r; // global row = b*L + l
                int b = mg >> 11, l = mg & (DL - 1);
                VT[((size_t)((b * DH + h) * DD + d)) * DL + l] =
                    f32_bf16_rne(acc[i][j][r] + bs);
            }
        }
}

// ---------------------------------------------------------------------------
// Attention: one block per (b, h, 16-row q tile). 128 threads = 4 waves.
// Phase 1: S = (Q·K^T)/8 into LDS (16 x 2048 f32).
// Phase 2: row softmax in LDS (exp in place, 1/rowsum to sInv).
// Phase 3: O = P·V via WMMA (P from LDS -> bf16, V from vT), scaled by sInv.
// ---------------------------------------------------------------------------
__global__ __launch_bounds__(128) void k_attention(const unsigned short* __restrict__ Qp,
                                                   const unsigned short* __restrict__ Kp,
                                                   const unsigned short* __restrict__ VT,
                                                   unsigned short* __restrict__ Op) {
    __shared__ float sS[16 * DL];       // 131072 B
    __shared__ float sPart[16][8];
    __shared__ float sInv[16];

    int tid = threadIdx.x;
    int wave = tid >> 5, lane = tid & 31;
    int mlane = lane & 15, hl = (lane >> 4) & 1;

    int bid = blockIdx.x;
    int qt = bid & 63;
    int bh = bid >> 6;
    int h = bh % DH;
    int b = bh / DH;
    int qbase = qt * 16;

    // Q fragments: rows qbase..qbase+15, d = 0..63 (two K-steps of 32)
    FragBF qa[2];
    {
        const unsigned short* qrow =
            Qp + ((size_t)(b * DLQ + qbase + mlane)) * DE + h * DD;
#pragma unroll
        for (int s = 0; s < 2; ++s) {
            qa[s].q[0] = *reinterpret_cast<const uint4*>(qrow + s * 32 + hl * 8);
            qa[s].q[1] = *reinterpret_cast<const uint4*>(qrow + s * 32 + hl * 8 + 16);
        }
    }

    // Phase 1: scores
    for (int kt = wave; kt < DL / 16; kt += 4) {
        v8f acc;
#pragma unroll
        for (int r = 0; r < 8; ++r) acc[r] = 0.0f;
        const unsigned short* krow =
            Kp + ((size_t)(b * DL + kt * 16 + mlane)) * DE + h * DD;
#pragma unroll
        for (int s = 0; s < 2; ++s) {
            FragBF fb;
            fb.q[0] = *reinterpret_cast<const uint4*>(krow + s * 32 + hl * 8);
            fb.q[1] = *reinterpret_cast<const uint4*>(krow + s * 32 + hl * 8 + 16);
            acc = wmma_bf16(qa[s].v, fb.v, acc);
        }
#pragma unroll
        for (int r = 0; r < 8; ++r)
            sS[(hl * 8 + r) * DL + kt * 16 + mlane] = acc[r] * 0.125f;
    }
    __syncthreads();

    // Phase 2: softmax (8 threads per row, 256 cols each)
    {
        int row = tid >> 3, seg = tid & 7;
        float* rp = sS + row * DL + seg * 256;
        float mx = -3.0e38f;
        for (int c = 0; c < 256; ++c) mx = fmaxf(mx, rp[c]);
        sPart[row][seg] = mx;
        __syncthreads();
        float rmx = sPart[row][0];
#pragma unroll
        for (int s = 1; s < 8; ++s) rmx = fmaxf(rmx, sPart[row][s]);
        float lsum = 0.0f;
        for (int c = 0; c < 256; ++c) {
            float e = __expf(rp[c] - rmx);
            rp[c] = e;
            lsum += e;
        }
        __syncthreads();
        sPart[row][seg] = lsum;
        __syncthreads();
        if (seg == 0) {
            float rs = 0.0f;
#pragma unroll
            for (int s = 0; s < 8; ++s) rs += sPart[row][s];
            sInv[row] = 1.0f / rs;
        }
        __syncthreads();
    }

    // Phase 3: O = P·V ; wave w owns d in [16w, 16w+16)
    v8f oacc;
#pragma unroll
    for (int r = 0; r < 8; ++r) oacc[r] = 0.0f;
    const unsigned short* vbase =
        VT + ((size_t)((b * DH + h) * DD + wave * 16 + mlane)) * DL;
    for (int kb = 0; kb < DL; kb += 32) {
        FragBF pa;
        const float* p0 = sS + mlane * DL + kb + hl * 8;
#pragma unroll
        for (int jj = 0; jj < 8; ++jj) {
            pa.h[jj] = f32_bf16_rne(p0[jj]);
            pa.h[8 + jj] = f32_bf16_rne(p0[16 + jj]);
        }
        FragBF vb;
        const unsigned short* vp = vbase + kb + hl * 8;
        vb.q[0] = *reinterpret_cast<const uint4*>(vp);
        vb.q[1] = *reinterpret_cast<const uint4*>(vp + 16);
        oacc = wmma_bf16(pa.v, vb.v, oacc);
    }
#pragma unroll
    for (int r = 0; r < 8; ++r) {
        int m = hl * 8 + r;
        float v = oacc[r] * sInv[m];
        Op[((size_t)(b * DLQ + qbase + m)) * DE + h * DD + wave * 16 + mlane] =
            f32_bf16_rne(v);
    }
}

// ---------------------------------------------------------------------------
// Final GEMM: out = a1 @ Wo^T + bo + state[:, :, ::2, :], permuted to [B,H,Lq,D]
// M = 8192. grid = (3, 128).
// ---------------------------------------------------------------------------
__global__ __launch_bounds__(128) void k_gemm_final(const unsigned short* __restrict__ A,
                                                    const unsigned short* __restrict__ W,
                                                    const float* __restrict__ bias,
                                                    const float* __restrict__ state,
                                                    float* __restrict__ out) {
    int wave = threadIdx.x >> 5, lane = threadIdx.x & 31;
    int mlane = lane & 15, hl = (lane >> 4) & 1;
    int m0 = blockIdx.y * 64;
    int n0 = blockIdx.x * 256 + wave * 64;

    v8f acc[4][4];
#pragma unroll
    for (int i = 0; i < 4; ++i)
#pragma unroll
        for (int j = 0; j < 4; ++j)
#pragma unroll
            for (int r = 0; r < 8; ++r) acc[i][j][r] = 0.0f;

    for (int kb = 0; kb < DE; kb += 32) {
        FragBF fa[4], fb[4];
#pragma unroll
        for (int i = 0; i < 4; ++i) {
            const unsigned short* p = A + ((size_t)(m0 + i * 16 + mlane)) * DE + kb + hl * 8;
            fa[i].q[0] = *reinterpret_cast<const uint4*>(p);
            fa[i].q[1] = *reinterpret_cast<const uint4*>(p + 16);
        }
#pragma unroll
        for (int j = 0; j < 4; ++j) {
            const unsigned short* p = W + ((size_t)(n0 + j * 16 + mlane)) * DE + kb + hl * 8;
            fb[j].q[0] = *reinterpret_cast<const uint4*>(p);
            fb[j].q[1] = *reinterpret_cast<const uint4*>(p + 16);
        }
#pragma unroll
        for (int i = 0; i < 4; ++i)
#pragma unroll
            for (int j = 0; j < 4; ++j)
                acc[i][j] = wmma_bf16(fa[i].v, fb[j].v, acc[i][j]);
    }

#pragma unroll
    for (int i = 0; i < 4; ++i)
#pragma unroll
        for (int j = 0; j < 4; ++j) {
            int n = n0 + j * 16 + mlane;
            int h = n >> 6, d = n & 63;
            float bs = bias[n];
#pragma unroll
            for (int r = 0; r < 8; ++r) {
                int mg = m0 + i * 16 + hl * 8 + r;   // b*Lq + lq
                int b = mg >> 10, lq = mg & (DLQ - 1);
                float res = state[(((size_t)(b * DH + h)) * DL + 2 * lq) * DD + d];
                out[(((size_t)(b * DH + h)) * DLQ + lq) * DD + d] =
                    acc[i][j][r] + bs + res;
            }
        }
}

// ---------------------------------------------------------------------------
// Host launcher
// ---------------------------------------------------------------------------
extern "C" void kernel_launch(void* const* d_in, const int* in_sizes, int n_in,
                              void* d_out, int out_size, void* d_ws, size_t ws_size,
                              hipStream_t stream) {
    (void)in_sizes; (void)n_in; (void)out_size; (void)ws_size;

    const float* state = (const float*)d_in[0];
    const float* Wi  = (const float*)d_in[1];  const float* bi  = (const float*)d_in[2];
    const float* Wq  = (const float*)d_in[3];  const float* bq  = (const float*)d_in[4];
    const float* Wk  = (const float*)d_in[5];  const float* bk  = (const float*)d_in[6];
    const float* Wv  = (const float*)d_in[7];  const float* bv  = (const float*)d_in[8];
    const float* Wao = (const float*)d_in[9];  const float* bao = (const float*)d_in[10];
    const float* Wo  = (const float*)d_in[11]; const float* bo  = (const float*)d_in[12];
    float* out = (float*)d_out;

    char* ws = (char*)d_ws;
    const size_t SZ_ROW  = (size_t)NROW * DE * 2;   // 25165824
    const size_t SZ_QROW = (size_t)NQROW * DE * 2;  // 12582912
    const size_t SZ_W    = (size_t)DE * DE * 2;     // 1179648

    unsigned short* x_bf  = (unsigned short*)(ws);                    // also o_bf
    unsigned short* xr_bf = (unsigned short*)(ws + SZ_ROW);
    unsigned short* k_bf  = (unsigned short*)(ws + 2 * SZ_ROW);
    unsigned short* vT_bf = (unsigned short*)(ws + 3 * SZ_ROW);
    unsigned short* q_bf  = (unsigned short*)(ws + 4 * SZ_ROW);       // also a1_bf
    char* wsw = ws + 4 * SZ_ROW + SZ_QROW;
    unsigned short* Wi_bf  = (unsigned short*)(wsw + 0 * SZ_W);
    unsigned short* Wk_bf  = (unsigned short*)(wsw + 1 * SZ_W);
    unsigned short* Wv_bf  = (unsigned short*)(wsw + 2 * SZ_W);
    unsigned short* Wao_bf = (unsigned short*)(wsw + 3 * SZ_W);
    unsigned short* Wo_bf  = (unsigned short*)(wsw + 4 * SZ_W);
    unsigned short* Wq_bf  = (unsigned short*)(wsw + 5 * SZ_W);       // 3*SZ_W
    unsigned short* o_bf  = x_bf;
    unsigned short* a1_bf = q_bf;

    // prep
    {
        int n = NROW * DE;
        k_permute_bf16<<<(n + 255) / 256, 256, 0, stream>>>(state, x_bf);
    }
    {
        int n = DE * DE;
        int g = (n + 255) / 256;
        k_cvt_bf16<<<g, 256, 0, stream>>>(Wi,  Wi_bf,  n);
        k_cvt_bf16<<<g, 256, 0, stream>>>(Wk,  Wk_bf,  n);
        k_cvt_bf16<<<g, 256, 0, stream>>>(Wv,  Wv_bf,  n);
        k_cvt_bf16<<<g, 256, 0, stream>>>(Wao, Wao_bf, n);
        k_cvt_bf16<<<g, 256, 0, stream>>>(Wo,  Wo_bf,  n);
        int n3 = 3 * DE * DE;
        k_transpose_wq<<<(n3 + 255) / 256, 256, 0, stream>>>(Wq, Wq_bf);
    }

    dim3 blk(128, 1, 1);
    dim3 gFull(3, NROW / 64, 1);    // (3, 256)
    dim3 gHalf(3, NQROW / 64, 1);   // (3, 128)

    // xr = relu(x @ Wi^T + bi)
    k_gemm_rm<<<gFull, blk, 0, stream>>>(x_bf, Wi_bf, bi, xr_bf, 1);
    // k = xr @ Wk^T + bk
    k_gemm_rm<<<gFull, blk, 0, stream>>>(xr_bf, Wk_bf, bk, k_bf, 0);
    // vT = (xr @ Wv^T + bv) transposed per head
    k_gemm_vT<<<gFull, blk, 0, stream>>>(xr_bf, Wv_bf, bv, vT_bf);
    // q = conv1d(xr, Wq, stride 2, pad 1) + bq
    k_gemm_conv<<<gHalf, blk, 0, stream>>>(xr_bf, Wq_bf, bq, q_bf);

    // attention
    k_attention<<<DB * DH * (DLQ / 16), blk, 0, stream>>>(q_bf, k_bf, vT_bf, o_bf);

    // a1 = relu(o @ Wao^T + bao)
    k_gemm_rm<<<gHalf, blk, 0, stream>>>(o_bf, Wao_bf, bao, a1_bf, 1);
    // out = a1 @ Wo^T + bo + residual, permuted
    k_gemm_final<<<gHalf, blk, 0, stream>>>(a1_bf, Wo_bf, bo, state, out);
}
